// SparseAttention_80625126081223
// MI455X (gfx1250) — matrix operand order repaired and checked by reading, not compiled, not run on hardware
//
#include <hip/hip_runtime.h>
#include <hip/hip_bf16.h>

// ---------------------------------------------------------------------------
// ESM-style attention with head-axis softmax, fused flash-style for MI455X.
// bf16 WMMA (v_wmma_f32_16x16x32_bf16) for all three GEMM stages.
// Fragment loads are batched ahead of back-to-back WMMA issue.
// ---------------------------------------------------------------------------

typedef __attribute__((ext_vector_type(16))) __bf16 v16bf;
typedef __attribute__((ext_vector_type(8)))  __bf16 v8bf;
typedef __attribute__((ext_vector_type(4)))  __bf16 v4bf;
typedef __attribute__((ext_vector_type(8)))  float  v8f;

#define S_LEN   3072
#define DMODEL  1280
#define NHEADS  20
#define HDIM    64
#define KTILES  (S_LEN / 64)   // 48

static __device__ inline v8f zero8() {
  v8f z;
#pragma unroll
  for (int i = 0; i < 8; ++i) z[i] = 0.0f;
  return z;
}

// ---- A fragment (16x32 bf16, row-major source) ----------------------------
// Lanes 0-15 -> M=lane, K base 0; lanes 16-31 -> M=lane-16, K base 8.
// Lane needs K = kb..kb+7 and kb+16..kb+23 -> two contiguous b128 loads.
static __device__ inline v16bf load_A_bf16(const __bf16* base, int ld) {
  const int lane = threadIdx.x & 31;
  const int m    = lane & 15;
  const int kb   = (lane < 16) ? 0 : 8;
  const __bf16* p = base + m * ld + kb;          // 16B-aligned
  v8bf lo = *(const v8bf*)p;                     // K = kb .. kb+7
  v8bf hi = *(const v8bf*)(p + 16);              // K = kb+16 .. kb+23
  v16bf a;
#pragma unroll
  for (int j = 0; j < 8; ++j) { a[j] = lo[j]; a[8 + j] = hi[j]; }
  return a;
}

// ---- B fragment (32x16 bf16, contraction contiguous in source rows) -------
// B[k][n] = src[n][k]. Lane n = lane&15; lanes 0-15 hold K=0-7,16-23; 16-31 K=8-15,24-31.
static __device__ inline v16bf load_B_bf16(const __bf16* base, int ld) {
  const int lane = threadIdx.x & 31;
  const int n    = lane & 15;
  const int kb   = (lane < 16) ? 0 : 8;
  const __bf16* p = base + n * ld + kb;          // 16B-aligned
  v8bf lo = *(const v8bf*)p;
  v8bf hi = *(const v8bf*)(p + 16);
  v16bf b;
#pragma unroll
  for (int j = 0; j < 8; ++j) { b[j] = lo[j]; b[j + 8] = hi[j]; }
  return b;
}

#define WMMA_BF16(A, B, C) \
  __builtin_amdgcn_wmma_f32_16x16x32_bf16(false, (A), false, (B), (short)0, (C), false, false)

// ---------------------------------------------------------------------------
// Kernel 0: one-shot fp32 -> bf16 conversion (4 elements / thread).
// ---------------------------------------------------------------------------
__global__ void cvt_bf16_kernel(const float* __restrict__ in,
                                __bf16* __restrict__ out, int n) {
  const int i4 = (blockIdx.x * blockDim.x + threadIdx.x) * 4;
  if (i4 >= n) return;
  const float4 f = *(const float4*)(in + i4);
  v4bf o;
  o[0] = (__bf16)f.x; o[1] = (__bf16)f.y; o[2] = (__bf16)f.z; o[3] = (__bf16)f.w;
  *(v4bf*)(out + i4) = o;
}

// ---------------------------------------------------------------------------
// Kernel 1: out[s,n] = (hs[s,:] . W[n,:] + bias[n]) * scale   (bf16 -> fp32)
// One wave computes a 32x64 tile (2 A-frags share 4 B-frags).
// grid = (D/64, S/32).
// ---------------------------------------------------------------------------
__global__ __launch_bounds__(32) void qkv_proj_kernel(
    const __bf16* __restrict__ hs, const __bf16* __restrict__ W,
    const float* __restrict__ bias, float* __restrict__ out, float scale) {
  const int n0 = blockIdx.x * 64;
  const int m0 = blockIdx.y * 32;

  v8f acc[2][4];
#pragma unroll
  for (int mi = 0; mi < 2; ++mi)
#pragma unroll
    for (int nt = 0; nt < 4; ++nt) acc[mi][nt] = zero8();

  for (int k0 = 0; k0 < DMODEL; k0 += 32) {
    // batch all fragment loads for this k-step
    v16bf a[2];
#pragma unroll
    for (int mi = 0; mi < 2; ++mi)
      a[mi] = load_A_bf16(hs + (size_t)(m0 + 16 * mi) * DMODEL + k0, DMODEL);
    v16bf b[4];
#pragma unroll
    for (int nt = 0; nt < 4; ++nt)
      b[nt] = load_B_bf16(W + (size_t)(n0 + nt * 16) * DMODEL + k0, DMODEL);
    // back-to-back WMMA on distinct accumulators
#pragma unroll
    for (int mi = 0; mi < 2; ++mi)
#pragma unroll
      for (int nt = 0; nt < 4; ++nt)
        acc[mi][nt] = WMMA_BF16(a[mi], b[nt], acc[mi][nt]);
  }

  const int lane = threadIdx.x & 31;
  const int m_hi = (lane < 16) ? 0 : 8;
  const int n_lo = lane & 15;
#pragma unroll
  for (int mi = 0; mi < 2; ++mi)
#pragma unroll
    for (int nt = 0; nt < 4; ++nt) {
      const int n = n0 + nt * 16 + n_lo;
      const float bn = bias[n];
#pragma unroll
      for (int r = 0; r < 8; ++r)
        out[(size_t)(m0 + 16 * mi + m_hi + r) * DMODEL + n] = (acc[mi][nt][r] + bn) * scale;
    }
}

// ---------------------------------------------------------------------------
// Kernel 2: full-width rotary on q,k (pairs (i, i+640)).
//   q,k -> head-major bf16 [H, S, 64]
//   v   -> transposed bf16 [DMODEL, S]   (per-head slices stay contiguous)
// ---------------------------------------------------------------------------
__global__ void rope_pack_kernel(const float* __restrict__ qt,
                                 const float* __restrict__ kt,
                                 const float* __restrict__ vt,
                                 __bf16* __restrict__ qh,
                                 __bf16* __restrict__ kh,
                                 __bf16* __restrict__ vT) {
  const int idx = blockIdx.x * blockDim.x + threadIdx.x;
  const int HALF = DMODEL / 2;  // 640
  if (idx >= S_LEN * HALF) return;
  const int s = idx / HALF;
  const int i = idx % HALF;

  const float inv = __expf(-9.210340372f * ((float)i / (float)HALF)); // 10000^(-i/640)
  const float ang = (float)s * inv;
  const float c = __cosf(ang), sn = __sinf(ang);

  const size_t row = (size_t)s * DMODEL;
  const int c1 = i, c2 = i + HALF;

  const float q1 = qt[row + c1], q2 = qt[row + c2];
  const float k1 = kt[row + c1], k2 = kt[row + c2];

  auto put_h = [&](__bf16* dst, int col, float v) {  // [H,S,64] head-major
    const int h = col >> 6, d = col & 63;
    dst[((size_t)h * S_LEN + s) * HDIM + d] = (__bf16)v;
  };
  put_h(qh, c1, q1 * c - q2 * sn);
  put_h(qh, c2, q2 * c + q1 * sn);
  put_h(kh, c1, k1 * c - k2 * sn);
  put_h(kh, c2, k2 * c + k1 * sn);
  // V transposed: vT[col][s]
  vT[(size_t)c1 * S_LEN + s] = (__bf16)vt[row + c1];
  vT[(size_t)c2 * S_LEN + s] = (__bf16)vt[row + c2];
}

// ---------------------------------------------------------------------------
// Kernel 3: fused attention with softmax over the HEAD axis.
// Block = 640 threads = 20 waves = 20 heads; one 16-query tile per block.
// Per 64-key tile: scores via WMMA, exp -> LDS, bank-parallel head-sum
// (stores 1/Z), p = exp * (1/Z) -> bf16 -> p.V via WMMA on transposed V.
// Output written flat in [H,S,hd] order (faithful to reference's raw view).
// ---------------------------------------------------------------------------
__global__ __launch_bounds__(640) void attn_kernel(
    const __bf16* __restrict__ qh, const __bf16* __restrict__ kh,
    const __bf16* __restrict__ vT, float* __restrict__ out) {
  __shared__ float  Zr[16][64];             // reciprocal of head-sum of exp
  __shared__ __bf16 P[NHEADS][16][64];      // per-head exp(scores)

  const int qt   = blockIdx.x;              // query tile (16 rows)
  const int tid  = threadIdx.x;
  const int h    = tid >> 5;                // head = wave
  const int lane = tid & 31;

  const int m_hi = (lane < 16) ? 0 : 8;
  const int n_lo = lane & 15;
  const int kbA  = (lane < 16) ? 0 : 8;

  // Q fragments (contraction over hd=64 -> 2 steps of 32); loop-invariant.
  const __bf16* qbase = qh + ((size_t)h * S_LEN + (size_t)qt * 16) * HDIM;
  v16bf aq[2];
  aq[0] = load_A_bf16(qbase + 0,  HDIM);
  aq[1] = load_A_bf16(qbase + 32, HDIM);

  v8f ctx[4];
#pragma unroll
  for (int nt = 0; nt < 4; ++nt) ctx[nt] = zero8();

  const __bf16* khead = kh + (size_t)h * S_LEN * HDIM;
  const __bf16* vhead = vT + (size_t)h * HDIM * S_LEN;

  for (int ktile = 0; ktile < KTILES; ++ktile) {
    // ---- scores s[16 q x 64 k] = q . k^T (no LDS dependence) ----
    const __bf16* kbase = khead + (size_t)ktile * 64 * HDIM;
    v16bf kb[2][4];
#pragma unroll
    for (int ks = 0; ks < 2; ++ks)
#pragma unroll
      for (int nt = 0; nt < 4; ++nt)
        kb[ks][nt] = load_B_bf16(kbase + (size_t)(nt * 16) * HDIM + ks * 32, HDIM);

    v8f sacc[4];
#pragma unroll
    for (int nt = 0; nt < 4; ++nt) sacc[nt] = zero8();
#pragma unroll
    for (int ks = 0; ks < 2; ++ks)
#pragma unroll
      for (int nt = 0; nt < 4; ++nt)
        sacc[nt] = WMMA_BF16(aq[ks], kb[ks][nt], sacc[nt]);

    if (ktile + 1 < KTILES) {                // prefetch next K/V tiles
      __builtin_prefetch(kbase + (size_t)64 * HDIM + lane * 64, 0, 1);
      __builtin_prefetch(vhead + (size_t)(lane * 2) * S_LEN + (ktile + 1) * 64, 0, 1);
    }

    __syncthreads();                         // prev-iter P/Zr reads done

    // ---- exp -> P ----
#pragma unroll
    for (int nt = 0; nt < 4; ++nt) {
      const int nloc = nt * 16 + n_lo;
#pragma unroll
      for (int r = 0; r < 8; ++r)
        P[h][m_hi + r][nloc] = (__bf16)__expf(sacc[nt][r]);
    }
    __syncthreads();                         // P complete

    // ---- head reduction: Zr = 1 / sum_h exp ----
    for (int z = tid; z < 16 * 64; z += 640) {
      const int m = z >> 6, n = z & 63;
      float s = 0.0f;
#pragma unroll
      for (int h2 = 0; h2 < NHEADS; ++h2) s += (float)P[h2][m][n];
      Zr[m][n] = 1.0f / s;
    }
    __syncthreads();                         // Zr ready

    // ---- ctx += (exp * Zr) . V  (contraction over 64 keys -> 2 steps) ----
    // pa fragments: vectorized LDS reads (two contiguous 8-element runs).
    const int m2 = lane & 15;
    const __bf16* prow = &P[h][m2][0];
    const float*  zrow = &Zr[m2][0];
    v16bf pa[2];
#pragma unroll
    for (int ks = 0; ks < 2; ++ks) {
      v8bf plo = *(const v8bf*)(prow + ks * 32 + kbA);        // 16B ds load
      v8bf phi = *(const v8bf*)(prow + ks * 32 + kbA + 16);
      v8f  zlo = *(const v8f*)(zrow + ks * 32 + kbA);         // 32B ds load
      v8f  zhi = *(const v8f*)(zrow + ks * 32 + kbA + 16);
#pragma unroll
      for (int j = 0; j < 8; ++j) {
        pa[ks][j]     = (__bf16)((float)plo[j] * zlo[j]);
        pa[ks][8 + j] = (__bf16)((float)phi[j] * zhi[j]);
      }
    }

    const int vcol0 = ktile * 64;
    v16bf bv[2][4];
#pragma unroll
    for (int ks = 0; ks < 2; ++ks)
#pragma unroll
      for (int nt = 0; nt < 4; ++nt)
        bv[ks][nt] = load_B_bf16(vhead + (size_t)(nt * 16) * S_LEN + vcol0 + ks * 32,
                                 S_LEN);
#pragma unroll
    for (int ks = 0; ks < 2; ++ks)
#pragma unroll
      for (int nt = 0; nt < 4; ++nt)
        ctx[nt] = WMMA_BF16(pa[ks], bv[ks][nt], ctx[nt]);
  }

  // ---- store: faithful raw reshape => flat [H, S, hd] order ----
#pragma unroll
  for (int nt = 0; nt < 4; ++nt)
#pragma unroll
    for (int r = 0; r < 8; ++r)
      out[((size_t)h * S_LEN + (size_t)qt * 16 + m_hi + r) * HDIM + nt * 16 + n_lo] =
          ctx[nt][r];
}

// ---------------------------------------------------------------------------
extern "C" void kernel_launch(void* const* d_in, const int* in_sizes, int n_in,
                              void* d_out, int out_size, void* d_ws, size_t ws_size,
                              hipStream_t stream) {
  (void)in_sizes; (void)n_in; (void)out_size; (void)ws_size;
  const float* hs = (const float*)d_in[0];
  const float* Wq = (const float*)d_in[1];
  const float* bq = (const float*)d_in[2];
  const float* Wk = (const float*)d_in[3];
  const float* bk = (const float*)d_in[4];
  const float* Wv = (const float*)d_in[5];
  const float* bv = (const float*)d_in[6];

  char* ws = (char*)d_ws;
  const size_t f32mat  = (size_t)S_LEN * DMODEL * sizeof(float);   // 15.7 MB
  const size_t bf16mat = (size_t)S_LEN * DMODEL * 2;               //  7.9 MB
  const size_t bf16w   = (size_t)DMODEL * DMODEL * 2;              //  3.3 MB
  size_t off = 0;
  float*  qtmp = (float*)(ws + off);  off += f32mat;
  float*  ktmp = (float*)(ws + off);  off += f32mat;
  float*  vtmp = (float*)(ws + off);  off += f32mat;
  __bf16* qhb  = (__bf16*)(ws + off); off += bf16mat;
  __bf16* khb  = (__bf16*)(ws + off); off += bf16mat;
  __bf16* vTb  = (__bf16*)(ws + off); off += bf16mat;
  __bf16* hsb  = (__bf16*)(ws + off); off += bf16mat;
  __bf16* Wqb  = (__bf16*)(ws + off); off += bf16w;
  __bf16* Wkb  = (__bf16*)(ws + off); off += bf16w;
  __bf16* Wvb  = (__bf16*)(ws + off); off += bf16w;

  // 0) one-shot bf16 staging of activations and weights
  const int nHS = S_LEN * DMODEL, nW = DMODEL * DMODEL;
  cvt_bf16_kernel<<<(nHS / 4 + 255) / 256, 256, 0, stream>>>(hs, hsb, nHS);
  cvt_bf16_kernel<<<(nW  / 4 + 255) / 256, 256, 0, stream>>>(Wq, Wqb, nW);
  cvt_bf16_kernel<<<(nW  / 4 + 255) / 256, 256, 0, stream>>>(Wk, Wkb, nW);
  cvt_bf16_kernel<<<(nW  / 4 + 255) / 256, 256, 0, stream>>>(Wv, Wvb, nW);

  // 1) QKV projections (q pre-scaled by hd^-0.5 = 1/8, matching reference)
  dim3 gproj(DMODEL / 64, S_LEN / 32);
  qkv_proj_kernel<<<gproj, 32, 0, stream>>>(hsb, Wqb, bq, qtmp, 0.125f);
  qkv_proj_kernel<<<gproj, 32, 0, stream>>>(hsb, Wkb, bk, ktmp, 1.0f);
  qkv_proj_kernel<<<gproj, 32, 0, stream>>>(hsb, Wvb, bv, vtmp, 1.0f);

  // 2) Rotary (full hidden width) + bf16 repack (V transposed)
  const int nrope = S_LEN * (DMODEL / 2);
  rope_pack_kernel<<<(nrope + 255) / 256, 256, 0, stream>>>(qtmp, ktmp, vtmp,
                                                            qhb, khb, vTb);

  // 3) Fused head-softmax attention
  attn_kernel<<<S_LEN / 16, 640, 0, stream>>>(qhb, khb, vTb, (float*)d_out);
}
